// TransformerWithKV_9139690405938
// MI455X (gfx1250) — matrix operand order repaired and checked by reading, not compiled, not run on hardware
//
#include <hip/hip_runtime.h>
#include <hip/hip_bf16.h>
#include <math.h>

// ---------------------------------------------------------------------------
// TransformerWithKV forward for MI455X (gfx1250, wave32, WMMA).
// All GEMMs use v_wmma_f32_16x16x32_bf16 (f32 accumulate). Bandwidth-bound
// workload (mem_keys 205MB dominates). GEMM A-tiles are streamed into LDS via
// CDNA5 async copies (global_load_async_to_lds_b32 + s_wait_asynccnt) with
// double buffering; sims GEMM normalizes keys on the fly in a single pass.
// ---------------------------------------------------------------------------

typedef __attribute__((ext_vector_type(16))) __bf16 v16bf;
typedef __attribute__((ext_vector_type(8)))  float  v8f;

#define WMMA_BF16(A, B, C) \
  __builtin_amdgcn_wmma_f32_16x16x32_bf16(false, (A), false, (B), (short)0, (C), false, false)

static constexpr int D_MODEL = 256;
static constexpr int N_HEAD  = 8;
static constexpr int DH      = 32;
static constexpr int D_FF    = 1024;
static constexpr int BATCH   = 64;
static constexpr int SEQ     = 256;
static constexpr int MEM     = 200000;
static constexpr int TOPK    = 32;
static constexpr int NTOK    = BATCH * SEQ;           // 16384
static constexpr int KTILES  = MEM / 16;              // 12500

// ---- workspace layout (floats) --------------------------------------------
static constexpr size_t OFS_X   = 0;                              // [NTOK,256]
static constexpr size_t OFS_QKV = OFS_X   + (size_t)NTOK * 256;   // [NTOK,768]
static constexpr size_t OFS_CTX = OFS_QKV + (size_t)NTOK * 768;   // [NTOK,256] (final-LN hidden reuses)
static constexpr size_t OFS_T2  = OFS_CTX + (size_t)NTOK * 256;   // [NTOK,256] (pooled/retr reuse)
static constexpr size_t OFS_T1  = OFS_T2  + (size_t)NTOK * 256;   // [NTOK,1024] (sims reuses: 12.8M <= 16.7M)
// total = 41,943,040 floats (~168 MB) required in d_ws

// ---- WMMA fragment builders (contiguous k order, same in A and B, wave32) ------------------
// A: 16x32 bf16, row r = lane&15; lanes 0-15 hold K 0..15,
//    lanes 16-31 hold K 16..31.
__device__ __forceinline__ v16bf load_a_frag(const float* src, int ld, float scale) {
  int lane = threadIdx.x & 31;
  int r  = lane & 15;
  int hb = (lane >> 4) * 16;
  const float* p = src + (size_t)r * ld;
  v16bf a;
#pragma unroll
  for (int j = 0; j < 8; ++j) {
    a[j]     = (__bf16)(p[hb + j]      * scale);
    a[j + 8] = (__bf16)(p[hb + 8 + j]  * scale);
  }
  return a;
}

// B: 32x16 bf16, col c = lane&15; lanes 0-15 hold K=0..15, lanes 16-31 K=16..31.
__device__ __forceinline__ v16bf load_b_frag(const float* src, int ld, float scale) {
  int lane = threadIdx.x & 31;
  int c  = lane & 15;
  int kb = (lane >> 4) * 16;
  v16bf b;
#pragma unroll
  for (int j = 0; j < 16; ++j)
    b[j] = (__bf16)(src[(size_t)(kb + j) * ld + c] * scale);
  return b;
}

// ---- CDNA5 async global->LDS copy of a 16x32 f32 tile (blockDim.x == 256) --
// Each thread issues 2 x b32 async transfers; tracked on ASYNCcnt, no VGPR
// data movement, no LOADcnt.  LDS byte address = low 32 bits of the generic
// pointer (ISA 10.2).
__device__ __forceinline__ void async_copy_tile_16x32(const float* gsrc, int ldg,
                                                      float* lds_buf) {
#pragma unroll
  for (int it = 0; it < 2; ++it) {
    int e  = threadIdx.x + it * 256;          // 0..511
    int r  = e >> 5, kk = e & 31;
    const float* gp = gsrc + (size_t)r * ldg + kk;
    unsigned int loff = (unsigned int)(size_t)(lds_buf + e);
    asm volatile("global_load_async_to_lds_b32 %0, %1, off"
                 :: "v"(loff), "v"(gp) : "memory");
  }
}

__device__ __forceinline__ void wait_async_all() {
  asm volatile("s_wait_asynccnt 0x0" ::: "memory");
}

// ---- embedding + sinusoidal positions -------------------------------------
__global__ void embed_kernel(const int* __restrict__ ids, const float* __restrict__ emb,
                             float* __restrict__ x) {
  int row = blockIdx.x;                 // b*SEQ + s
  int d   = threadIdx.x;
  int s   = row & (SEQ - 1);
  int tok = ids[row];
  int i2  = (d >> 1) * 2;
  float freq = __expf(-(float)i2 * (9.210340371976184f / (float)D_MODEL));
  float ang  = (float)s * freq;
  float pe   = (d & 1) ? cosf(ang) : sinf(ang);
  x[(size_t)row * D_MODEL + d] = emb[(size_t)tok * D_MODEL + d] * 16.0f + pe;
}

// ---- generic bf16-WMMA GEMM: C[M,N] = A[M,K] @ W[K,N] (+bias, +gelu) -------
// block: 256 thr (8 waves) -> 16 rows x 128 cols.  A tile double-buffered in
// LDS via async copies; one barrier per 32-K chunk.
__global__ void gemm_bf16(const float* __restrict__ A, const float* __restrict__ W,
                          const float* __restrict__ bias, float* __restrict__ C,
                          int M, int N, int K, int fuse /*0 none,1 bias,2 bias+gelu*/) {
  __shared__ float a_lds[2][16 * 32];
  int wave = threadIdx.x >> 5;
  int lane = threadIdx.x & 31;
  int row0 = blockIdx.y * 16;
  int col0 = (blockIdx.x * 8 + wave) * 16;
  const float* arow = A + (size_t)row0 * K;

  async_copy_tile_16x32(arow, K, a_lds[0]);

  v8f acc = {};
  int buf = 0;
  for (int kc = 0; kc < K; kc += 32) {
    wait_async_all();          // own async writes done
    __syncthreads();           // => every wave's writes done & reads of buf^1 retired
    if (kc + 32 < K)
      async_copy_tile_16x32(arow + kc + 32, K, a_lds[buf ^ 1]);
    v16bf a = load_a_frag(a_lds[buf], 32, 1.0f);
    v16bf b = load_b_frag(W + (size_t)kc * N + col0, N, 1.0f);
    acc = WMMA_BF16(a, b, acc);
    buf ^= 1;
  }
  int n  = col0 + (lane & 15);
  float bs = (fuse > 0) ? bias[n] : 0.0f;
  int mb = row0 + (lane >> 4) * 8;
#pragma unroll
  for (int v = 0; v < 8; ++v) {
    float val = acc[v] + bs;
    if (fuse == 2) val = 0.5f * val * (1.0f + erff(val * 0.70710678118654752f));
    C[(size_t)(mb + v) * N + n] = val;
  }
}

// ---- attention: one wave per (qtile, head, batch); scores strip in LDS -----
__global__ void attn_kernel(const float* __restrict__ qkv, float* __restrict__ ctx) {
  __shared__ float sc[16][SEQ];     // 16 KB
  int qt = blockIdx.x, h = blockIdx.y, b = blockIdx.z;
  int lane = threadIdx.x & 31;
  int r = lane & 15, half = lane >> 4;
  const float* base = qkv + (size_t)b * SEQ * 768;
  const float  qscale = 0.17677669529663689f;      // 1/sqrt(32)

  // Q fragment (pre-scaled)
  v16bf aq;
  {
    const float* qp = base + (size_t)(qt * 16 + r) * 768 + h * DH;
    int hb = half * 16;
#pragma unroll
    for (int j = 0; j < 8; ++j) {
      aq[j]     = (__bf16)(qp[hb + j]      * qscale);
      aq[j + 8] = (__bf16)(qp[hb + 8 + j]  * qscale);
    }
  }
  // scores = Q @ K^T
  for (int kt = 0; kt < SEQ / 16; ++kt) {
    v16bf bk;
    {
      const float* kp = base + (size_t)(kt * 16 + (lane & 15)) * 768 + D_MODEL + h * DH;
      int kb = half * 16;
#pragma unroll
      for (int j = 0; j < 16; ++j) bk[j] = (__bf16)kp[kb + j];
    }
    v8f s = {};
    s = WMMA_BF16(aq, bk, s);
#pragma unroll
    for (int v = 0; v < 8; ++v) sc[half * 8 + v][kt * 16 + (lane & 15)] = s[v];
  }
  __syncthreads();
  // softmax (one lane per row)
  if (lane < 16) {
    float mx = -3.4e38f;
    for (int c = 0; c < SEQ; ++c) mx = fmaxf(mx, sc[lane][c]);
    float sum = 0.0f;
    for (int c = 0; c < SEQ; ++c) { float e = __expf(sc[lane][c] - mx); sc[lane][c] = e; sum += e; }
    float inv = 1.0f / sum;
    for (int c = 0; c < SEQ; ++c) sc[lane][c] *= inv;
  }
  __syncthreads();
  // ctx = P @ V  (two 16-col halves of dh=32)
  for (int vt = 0; vt < 2; ++vt) {
    v8f acc = {};
    for (int ch = 0; ch < SEQ / 32; ++ch) {
      v16bf ap;
      {
        int hb = half * 16;
#pragma unroll
        for (int j = 0; j < 8; ++j) {
          ap[j]     = (__bf16)sc[r][ch * 32 + hb + j];
          ap[j + 8] = (__bf16)sc[r][ch * 32 + hb + 8 + j];
        }
      }
      v16bf bv = load_b_frag(base + (size_t)(ch * 32) * 768 + 2 * D_MODEL + h * DH + vt * 16, 768, 1.0f);
      acc = WMMA_BF16(ap, bv, acc);
    }
#pragma unroll
    for (int v = 0; v < 8; ++v)
      ctx[((size_t)b * SEQ + qt * 16 + half * 8 + v) * D_MODEL + h * DH + vt * 16 + (lane & 15)] = acc[v];
  }
}

// ---- layernorm (optionally + residual); one block per row ------------------
__global__ void ln_kernel(const float* __restrict__ x, const float* __restrict__ res,
                          const float* __restrict__ sc, const float* __restrict__ bs,
                          float* __restrict__ out) {
  __shared__ float red[D_MODEL];
  int row = blockIdx.x, d = threadIdx.x;
  float v = x[(size_t)row * D_MODEL + d] + (res ? res[(size_t)row * D_MODEL + d] : 0.0f);
  red[d] = v; __syncthreads();
  for (int off = 128; off > 0; off >>= 1) { if (d < off) red[d] += red[d + off]; __syncthreads(); }
  float mu = red[0] * (1.0f / D_MODEL); __syncthreads();
  float c = v - mu;
  red[d] = c * c; __syncthreads();
  for (int off = 128; off > 0; off >>= 1) { if (d < off) red[d] += red[d + off]; __syncthreads(); }
  float var = red[0] * (1.0f / D_MODEL);
  out[(size_t)row * D_MODEL + d] = c * rsqrtf(var + 1e-5f) * sc[d] + bs[d];
}

// ---- masked mean pool + normalized copy ------------------------------------
__global__ void pool_kernel(const float* __restrict__ hidden, const float* __restrict__ mask,
                            float* __restrict__ pooled, float* __restrict__ pooledn) {
  __shared__ float red[D_MODEL];
  int b = blockIdx.x, d = threadIdx.x;
  float msum = 0.0f, acc = 0.0f;
  for (int s = 0; s < SEQ; ++s) {
    float mm = mask[b * SEQ + s];
    msum += mm;
    acc  += hidden[((size_t)b * SEQ + s) * D_MODEL + d] * mm;
  }
  float p = acc / (msum + 1e-8f);
  pooled[b * D_MODEL + d] = p;
  red[d] = p * p; __syncthreads();
  for (int off = 128; off > 0; off >>= 1) { if (d < off) red[d] += red[d + off]; __syncthreads(); }
  float nrm = sqrtf(red[0]);
  pooledn[b * D_MODEL + d] = p / fmaxf(nrm, 1e-8f);
}

// ---- cosine sims: pooled_n[64,256] x mem_keys^T -> sims[64, MEM] ----------
// one wave per 16-key tile; key norms computed on the fly (single key pass).
__global__ void sims_kernel(const float* __restrict__ pooledn, const float* __restrict__ keys,
                            float* __restrict__ sims) {
  int wave = threadIdx.x >> 5, lane = threadIdx.x & 31;
  int t = blockIdx.x * 8 + wave;
  if (t >= KTILES) return;                        // wave-uniform
  int key0 = t * 16;
  int kidx = key0 + (lane & 15);
  int half = lane >> 4;

  if (t + 8 < KTILES)
    __builtin_prefetch(keys + (size_t)(kidx + 128) * D_MODEL, 0, 3);   // global_prefetch_b8

  // per-lane key ||k||: lane covers its key's dims [half*128, half*128+128)
  float ss = 0.0f;
  {
    const float* kr = keys + (size_t)kidx * D_MODEL + half * 128;
    for (int i = 0; i < 128; ++i) { float xv = kr[i]; ss += xv * xv; }
  }
  ss += __shfl_xor(ss, 16, 32);
  float inv = 1.0f / fmaxf(sqrtf(ss), 1e-8f);

  const float* kp = keys + (size_t)kidx * D_MODEL;   // lane's key row (B column)
  int kb = half * 16;
  for (int qt = 0; qt < 4; ++qt) {
    v8f acc = {};
    for (int kc = 0; kc < D_MODEL; kc += 32) {
      v16bf a = load_a_frag(pooledn + (size_t)(qt * 16) * D_MODEL + kc, D_MODEL, 1.0f);
      v16bf bk;
#pragma unroll
      for (int j = 0; j < 16; ++j) bk[j] = (__bf16)(kp[kc + kb + j] * inv);
      acc = WMMA_BF16(a, bk, acc);
    }
#pragma unroll
    for (int v = 0; v < 8; ++v)
      sims[(size_t)(qt * 16 + half * 8 + v) * MEM + key0 + (lane & 15)] = acc[v];
  }
}

// ---- exact top-32 + weighted aggregate; one block (128 thr) per batch row --
__global__ void topk_kernel(const float* __restrict__ sims, const float* __restrict__ conf,
                            const float* __restrict__ vals, float* __restrict__ retr) {
  __shared__ float cv[128 * TOPK];
  __shared__ int   ci[128 * TOPK];
  __shared__ float rv[128]; __shared__ int ri[128];
  __shared__ float selv[TOPK]; __shared__ int seli[TOPK];
  __shared__ float w[TOPK]; __shared__ float wsum_sh;
  int b = blockIdx.x, tid = threadIdx.x;
  const float* srow = sims + (size_t)b * MEM;

  float tv[TOPK]; int ti[TOPK];
#pragma unroll
  for (int i = 0; i < TOPK; ++i) { tv[i] = -3.4e38f; ti[i] = 0; }
  float vmin = -3.4e38f; int pmin = 0;
  for (int m = tid; m < MEM; m += 128) {
    float vv = srow[m];
    if (vv > vmin) {
      tv[pmin] = vv; ti[pmin] = m;
      vmin = tv[0]; pmin = 0;
      for (int i = 1; i < TOPK; ++i) if (tv[i] < vmin) { vmin = tv[i]; pmin = i; }
    }
  }
#pragma unroll
  for (int i = 0; i < TOPK; ++i) { cv[tid * TOPK + i] = tv[i]; ci[tid * TOPK + i] = ti[i]; }
  __syncthreads();

  for (int s = 0; s < TOPK; ++s) {
    float best = -3.4e38f; int bi = 0;
    for (int j = tid; j < 128 * TOPK; j += 128)
      if (cv[j] > best) { best = cv[j]; bi = j; }
    rv[tid] = best; ri[tid] = bi; __syncthreads();
    for (int off = 64; off > 0; off >>= 1) {
      if (tid < off && rv[tid + off] > rv[tid]) { rv[tid] = rv[tid + off]; ri[tid] = ri[tid + off]; }
      __syncthreads();
    }
    if (tid == 0) { int e = ri[0]; selv[s] = cv[e]; seli[s] = ci[e]; cv[e] = -3.4e38f; }
    __syncthreads();
  }
  if (tid < TOPK) w[tid] = fmaxf(conf[seli[tid]], 1e-4f) * selv[tid];
  __syncthreads();
  if (tid == 0) { float sw = 0.0f; for (int i = 0; i < TOPK; ++i) sw += w[i]; wsum_sh = sw + 1e-8f; }
  __syncthreads();
  for (int dd = tid; dd < D_MODEL; dd += 128) {
    float agg = 0.0f;
    for (int i = 0; i < TOPK; ++i) agg += w[i] * vals[(size_t)seli[i] * D_MODEL + dd];
    retr[b * D_MODEL + dd] = agg / wsum_sh;
  }
}

// ---- gate MLP + fuse + classifier; one block per batch ---------------------
__global__ void gate_cls_kernel(const float* __restrict__ pooled, const float* __restrict__ retr,
                                const float* __restrict__ gw1, const float* __restrict__ gb1,
                                const float* __restrict__ gw2, const float* __restrict__ gb2,
                                const float* __restrict__ cw, const float* __restrict__ cb,
                                float* __restrict__ out) {
  __shared__ float red[D_MODEL];
  __shared__ float fused[D_MODEL];
  __shared__ float gsh;
  int b = blockIdx.x, d = threadIdx.x;
  const float* p = pooled + b * D_MODEL;
  float acc = gb1[d];
  for (int k = 0; k < D_MODEL; ++k) acc += p[k] * gw1[k * D_MODEL + d];
  float hid = fmaxf(acc, 0.0f);
  red[d] = hid * gw2[d]; __syncthreads();
  for (int off = 128; off > 0; off >>= 1) { if (d < off) red[d] += red[d + off]; __syncthreads(); }
  if (d == 0) gsh = 1.0f / (1.0f + __expf(-(red[0] + gb2[0])));
  __syncthreads();
  float g = gsh;
  fused[d] = g * retr[b * D_MODEL + d] + (1.0f - g) * p[d];
  __syncthreads();
  if (d < 16) {
    float lg = cb[d];
    for (int k = 0; k < D_MODEL; ++k) lg += fused[k] * cw[k * 16 + d];
    out[b * 16 + d] = lg;
  }
}

// ---------------------------------------------------------------------------
extern "C" void kernel_launch(void* const* d_in, const int* in_sizes, int n_in,
                              void* d_out, int out_size, void* d_ws, size_t ws_size,
                              hipStream_t stream) {
  const int*   ids   = (const int*)  d_in[0];
  const float* mask  = (const float*)d_in[1];
  // d_in[2] = top_k (==32, hardcoded)
  const float* emb   = (const float*)d_in[3];
  const float* qkv_w = (const float*)d_in[4];
  const float* qkv_b = (const float*)d_in[5];
  const float* out_w = (const float*)d_in[6];
  const float* out_b = (const float*)d_in[7];
  const float* ln1_s = (const float*)d_in[8];
  const float* ln1_b = (const float*)d_in[9];
  const float* ff_w1 = (const float*)d_in[10];
  const float* ff_b1 = (const float*)d_in[11];
  const float* ff_w2 = (const float*)d_in[12];
  const float* ff_b2 = (const float*)d_in[13];
  const float* ln2_s = (const float*)d_in[14];
  const float* ln2_b = (const float*)d_in[15];
  const float* lnf_s = (const float*)d_in[16];
  const float* lnf_b = (const float*)d_in[17];
  const float* gw1   = (const float*)d_in[18];
  const float* gb1   = (const float*)d_in[19];
  const float* gw2   = (const float*)d_in[20];
  const float* gb2   = (const float*)d_in[21];
  const float* cw    = (const float*)d_in[22];
  const float* cb    = (const float*)d_in[23];
  const float* mkeys = (const float*)d_in[24];
  const float* mvals = (const float*)d_in[25];
  const float* mconf = (const float*)d_in[26];

  float* ws  = (float*)d_ws;
  float* X   = ws + OFS_X;
  float* QKV = ws + OFS_QKV;
  float* CTX = ws + OFS_CTX;
  float* T2  = ws + OFS_T2;
  float* T1  = ws + OFS_T1;
  float* HID     = CTX;                     // reuse after layers
  float* POOLED  = T2;
  float* POOLEDN = T2 + BATCH * D_MODEL;
  float* RETR    = T2 + 2 * BATCH * D_MODEL;
  float* SIMS    = T1;                      // 12.8M floats fits in 16.7M region

  embed_kernel<<<NTOK, D_MODEL, 0, stream>>>(ids, emb, X);

  for (int l = 0; l < 2; ++l) {
    gemm_bf16<<<dim3(768 / 128, NTOK / 16), 256, 0, stream>>>(
        X, qkv_w + (size_t)l * D_MODEL * 768, qkv_b + (size_t)l * 768, QKV,
        NTOK, 768, D_MODEL, 1);
    attn_kernel<<<dim3(SEQ / 16, N_HEAD, BATCH), 32, 0, stream>>>(QKV, CTX);
    gemm_bf16<<<dim3(D_MODEL / 128, NTOK / 16), 256, 0, stream>>>(
        CTX, out_w + (size_t)l * D_MODEL * D_MODEL, out_b + (size_t)l * D_MODEL, T2,
        NTOK, D_MODEL, D_MODEL, 1);
    ln_kernel<<<NTOK, D_MODEL, 0, stream>>>(X, T2, ln1_s + l * D_MODEL, ln1_b + l * D_MODEL, X);
    gemm_bf16<<<dim3(D_FF / 128, NTOK / 16), 256, 0, stream>>>(
        X, ff_w1 + (size_t)l * D_MODEL * D_FF, ff_b1 + (size_t)l * D_FF, T1,
        NTOK, D_FF, D_MODEL, 2 /* bias+gelu */);
    gemm_bf16<<<dim3(D_MODEL / 128, NTOK / 16), 256, 0, stream>>>(
        T1, ff_w2 + (size_t)l * D_FF * D_MODEL, ff_b2 + (size_t)l * D_MODEL, T2,
        NTOK, D_MODEL, D_FF, 1);
    ln_kernel<<<NTOK, D_MODEL, 0, stream>>>(X, T2, ln2_s + l * D_MODEL, ln2_b + l * D_MODEL, X);
  }

  ln_kernel<<<NTOK, D_MODEL, 0, stream>>>(X, nullptr, lnf_s, lnf_b, HID);
  pool_kernel<<<BATCH, D_MODEL, 0, stream>>>(HID, mask, POOLED, POOLEDN);
  sims_kernel<<<(KTILES + 7) / 8, 256, 0, stream>>>(POOLEDN, mkeys, SIMS);
  topk_kernel<<<BATCH, 128, 0, stream>>>(SIMS, mconf, mvals, RETR);
  gate_cls_kernel<<<BATCH, D_MODEL, 0, stream>>>(POOLED, RETR, gw1, gb1, gw2, gb2, cw, cb,
                                                 (float*)d_out);
}
